// Model_55705725829628
// MI455X (gfx1250) — matrix operand order repaired and checked by reading, not compiled, not run on hardware
//
#include <hip/hip_runtime.h>
#include <cstdint>

#define DEV __device__ __forceinline__

typedef __attribute__((ext_vector_type(16))) _Float16 v16h;
typedef __attribute__((ext_vector_type(8)))  _Float16 v8h;
typedef __attribute__((ext_vector_type(8)))  float    v8f;

namespace {
constexpr int N_ = 300, B_ = 32, T_ = 168, H_ = 128, NH_ = 8;
constexpr int SEM_IN = 128, SEM_DIM = 168, CH = 64, HOR = 12;
constexpr int FN = 336, FNP = 352;   // node feature dim, padded to mult of 32
constexpr int G3 = 384;              // 3*H
constexpr int TP = 176;              // T padded to mult of 16
constexpr int NP = 304;              // N padded to mult of 16
constexpr int JP = 320;              // N padded to mult of 32 (K pad for agg)
constexpr int QK = 256;              // q|k output cols
}

DEV v16h cat16(v8h a, v8h b) {
  return __builtin_shufflevector(a, b, 0,1,2,3,4,5,6,7,8,9,10,11,12,13,14,15);
}
// A fragment: 16x32 f16 tile, row-major LDS source, row stride ld (halves).
DEV v16h frag_a(const _Float16* src, int ld) {
  int l = threadIdx.x & 31;
  const _Float16* p = src + (l & 15) * ld + ((l >> 4) << 3);
  return cat16(*(const v8h*)p, *(const v8h*)(p + 16));
}
// B fragment: 32x16 tile of B[k][n], sourced from LDS storing Bt[n][k] (n-major).
DEV v16h frag_b(const _Float16* src, int ld) {
  int l = threadIdx.x & 31;
  const _Float16* p = src + (l & 15) * ld + ((l >> 4) << 3);
  return cat16(*(const v8h*)p, *(const v8h*)(p + 16));
}
DEV v8f wmma16(v16h a, v16h b, v8f c) {
  return __builtin_amdgcn_wmma_f32_16x16x32_f16(false, a, false, b, (short)0, c,
                                                false, false);
}
// fast sigmoid/tanh: CDNA5 has TANH/RCP/EXP as native TRANS ops
DEV float sigm(float x) { return __builtin_amdgcn_rcpf(1.f + __expf(-x)); }
DEV float ftanh(float x) {
#if __has_builtin(__builtin_amdgcn_tanhf)
  return __builtin_amdgcn_tanhf(x);
#elif __has_builtin(__builtin_amdgcn_tanh_f32)
  return __builtin_amdgcn_tanh_f32(x);
#else
  return 1.f - 2.f * __builtin_amdgcn_rcpf(__expf(2.f * x) + 1.f);
#endif
}

// gfx1250 async global->LDS copy (16B per lane), tracked by ASYNCcnt
DEV void async_ld_b128(uint32_t lds_addr, uint32_t goff, uint64_t sbase) {
  asm volatile("global_load_async_to_lds_b128 %0, %1, %2"
               :: "v"(lds_addr), "v"(goff), "s"(sbase) : "memory");
}
DEV void wait_async0() { asm volatile("s_wait_asynccnt 0x0" ::: "memory"); }

// ---------------------------------------------------------------- GRU (fused seq1)
__global__ void gru_kernel(const float* __restrict__ x,
    const float* __restrict__ w_ih, const float* __restrict__ w_hh,
    const float* __restrict__ b_ih, const float* __restrict__ b_hh,
    const float* __restrict__ s1w, const float* __restrict__ s1b,
    const float* __restrict__ s2w, const float* __restrict__ s2b,
    _Float16* __restrict__ gru_out, float* __restrict__ hid32)
{
  extern __shared__ char smem[];
  _Float16* Wih = (_Float16*)smem;            // [384][64]  (n-major: [out][k])
  _Float16* Whh = Wih + G3 * 64;              // [384][128]
  _Float16* Ht  = Whh + G3 * H_;              // [32][128] h (f16 copy)
  _Float16* Xt  = Ht + B_ * H_;               // [32][64]  seq1 output this step
  float* H32 = (float*)(Xt + B_ * 64);        // [32][128]
  float* GX  = H32 + B_ * H_;                 // [32][384] x-gates
  float* GH  = GX + B_ * G3;                  // [32][384] h-gates
  float* S1W = GH + B_ * G3; float* S1B = S1W + 32;
  float* S2W = S1B + 32;     float* S2B = S2W + 32 * 64;
  float* BIH = S2B + 64;     float* BHH = BIH + G3;

  const int n = blockIdx.x, tid = threadIdx.x;
  for (int e = tid; e < G3 * 64; e += 256) Wih[e] = (_Float16)w_ih[(size_t)n * G3 * 64 + e];
  for (int e = tid; e < G3 * H_; e += 256) Whh[e] = (_Float16)w_hh[(size_t)n * G3 * H_ + e];
  for (int e = tid; e < G3; e += 256) { BIH[e] = b_ih[n * G3 + e]; BHH[e] = b_hh[n * G3 + e]; }
  for (int e = tid; e < 32; e += 256) { S1W[e] = s1w[e]; S1B[e] = s1b[e]; }
  for (int e = tid; e < 32 * 64; e += 256) S2W[e] = s2w[e];
  for (int e = tid; e < 64; e += 256) S2B[e] = s2b[e];
  for (int e = tid; e < B_ * H_; e += 256) { H32[e] = 0.f; Ht[e] = (_Float16)0.f; }
  __syncthreads();

  const int w = tid >> 5, lane = tid & 31;
  for (int t = 0; t < T_; ++t) {
    { // fused seq1 MLP: x scalar -> relu 32 -> relu 64
      int b = tid >> 3, j = tid & 7;
      float xv = x[((size_t)b * T_ + t) * N_ + n];
      if ((tid & 7) == 0 && t + 1 < T_)                    // prefetch next step
        __builtin_prefetch(&x[((size_t)b * T_ + t + 1) * N_ + n], 0, 1);
      float h1[32];
      #pragma unroll
      for (int i = 0; i < 32; ++i) { float v = xv * S1W[i] + S1B[i]; h1[i] = v > 0.f ? v : 0.f; }
      #pragma unroll
      for (int cc = 0; cc < 8; ++cc) {
        int c = j * 8 + cc; float acc = S2B[c];
        #pragma unroll
        for (int i = 0; i < 32; ++i) acc += h1[i] * S2W[i * 64 + c];
        Xt[b * 64 + c] = (_Float16)(acc > 0.f ? acc : 0.f);
      }
    }
    __syncthreads();
    // gates: [32,384] = Xt@Wih^T and Ht@Whh^T ; 48 tiles over 8 waves
    for (int tt = w; tt < 48; tt += 8) {
      int mt = tt & 1, nt = tt >> 1;
      v8f ax = {}; v8f ah = {};
      #pragma unroll
      for (int kk = 0; kk < 2; ++kk)
        ax = wmma16(frag_a(Xt + mt * 16 * 64 + kk * 32, 64),
                    frag_b(Wih + nt * 16 * 64 + kk * 32, 64), ax);
      #pragma unroll
      for (int kk = 0; kk < 4; ++kk)
        ah = wmma16(frag_a(Ht + mt * 16 * H_ + kk * 32, H_),
                    frag_b(Whh + nt * 16 * H_ + kk * 32, H_), ah);
      int col = nt * 16 + (lane & 15), rb = mt * 16 + ((lane >> 4) << 3);
      #pragma unroll
      for (int r = 0; r < 8; ++r) { GX[(rb + r) * G3 + col] = ax[r]; GH[(rb + r) * G3 + col] = ah[r]; }
    }
    __syncthreads();
    for (int e = tid; e < B_ * H_; e += 256) {
      int b = e >> 7, c = e & 127;
      float r  = sigm(GX[b * G3 + c]       + BIH[c]       + GH[b * G3 + c]       + BHH[c]);
      float z  = sigm(GX[b * G3 + 128 + c] + BIH[128 + c] + GH[b * G3 + 128 + c] + BHH[128 + c]);
      float nn = ftanh(GX[b * G3 + 256 + c] + BIH[256 + c] +
                       r * (GH[b * G3 + 256 + c] + BHH[256 + c]));
      float hv = (1.f - z) * nn + z * H32[e];
      H32[e] = hv; Ht[e] = (_Float16)hv;
      gru_out[(((size_t)n * B_ + b) * T_ + t) * H_ + c] = (_Float16)hv;
    }
    __syncthreads();
  }
  for (int e = tid; e < B_ * H_; e += 256) hid32[(size_t)n * B_ * H_ + e] = H32[e];
}

// ------------------------------------------------------- temporal attention + LN
__global__ void attn_time_kernel(const _Float16* __restrict__ gru_out,
    const float* __restrict__ hid32, const float* __restrict__ attn_w,
    const float* __restrict__ attn_b, const float* __restrict__ attn_v,
    const float* __restrict__ ln_g, const float* __restrict__ ln_b,
    _Float16* __restrict__ wres)
{
  extern __shared__ char smem[];
  _Float16* AW = (_Float16*)smem;          // [128][256]  AW[j][k] = attn_w[k][j]
  _Float16* Gt = AW + H_ * 256;            // [176][128]  gru_out rows, T padded
  float* HT = (float*)(Gt + TP * H_);      // [128] hid-term + bias
  float* HF = HT + H_;                     // [128] hid
  float* VV = HF + H_;                     // [128] attn_v
  float* SC = VV + H_;                     // [176] scores
  float* RD = SC + TP;                     // [4]
  const int n = blockIdx.x, b = blockIdx.y, tid = threadIdx.x;
  const size_t base = (size_t)n * B_ + b;
  // async global->LDS tile fill of gru_out[n,b]: T_*H_ halves = 2688 x 16B
  {
    uint64_t sbase = (uint64_t)(uintptr_t)gru_out;
    uint32_t g0 = (uint32_t)(base * T_ * H_ * 2);
    uint32_t l0 = (uint32_t)(uintptr_t)Gt;
    for (int c = tid; c < (T_ * H_) / 8; c += 128)
      async_ld_b128(l0 + c * 16, g0 + c * 16, sbase);
  }
  for (int e = tid; e < H_ * 256; e += 128) {
    int j = e >> 8, k = e & 255;
    AW[e] = (_Float16)attn_w[k * H_ + j];
  }
  for (int e = tid + T_ * H_; e < TP * H_; e += 128) Gt[e] = (_Float16)0.f;
  HF[tid] = hid32[base * H_ + tid];
  VV[tid] = attn_v[tid];
  for (int e = tid; e < TP; e += 128) SC[e] = 0.f;
  wait_async0();
  __syncthreads();
  { // hid part of the concat GEMM (rank-constant over t)
    float acc = attn_b[tid];
    for (int k = 0; k < H_; ++k) acc += HF[k] * (float)AW[tid * 256 + k];
    HT[tid] = acc;
  }
  __syncthreads();
  const int w = tid >> 5, lane = tid & 31;
  for (int tt = w; tt < (TP / 16) * (H_ / 16); tt += 4) {
    int mt = tt % (TP / 16), nt = tt / (TP / 16);
    v8f acc = {};
    #pragma unroll
    for (int kk = 0; kk < 4; ++kk)
      acc = wmma16(frag_a(Gt + mt * 16 * H_ + kk * 32, H_),
                   frag_b(AW + nt * 16 * 256 + 128 + kk * 32, 256), acc);
    int col = nt * 16 + (lane & 15), rb = mt * 16 + ((lane >> 4) << 3);
    float vj = VV[col], hj = HT[col];
    #pragma unroll
    for (int r = 0; r < 8; ++r) {
      float e = ftanh(acc[r] + hj);
      atomicAdd(&SC[rb + r], e * vj);           // ds_add_f32
    }
  }
  __syncthreads();
  if (tid == 0) { float m = -1e30f; for (int t = 0; t < T_; ++t) m = fmaxf(m, SC[t]); RD[0] = m; }
  __syncthreads();
  for (int t = tid; t < T_; t += 128) SC[t] = __expf(SC[t] - RD[0]);
  __syncthreads();
  if (tid == 0) { float s = 0.f; for (int t = 0; t < T_; ++t) s += SC[t]; RD[1] = 1.f / s; }
  __syncthreads();
  float acc = 0.f;
  for (int t = 0; t < T_; ++t) acc += SC[t] * (float)Gt[t * H_ + tid];
  float z = acc * RD[1] + HF[tid];
  __syncthreads();
  SC[tid] = z;                                   // reuse as LN scratch
  __syncthreads();
  if (tid == 0) {
    float mu = 0.f; for (int c = 0; c < H_; ++c) mu += SC[c]; mu /= H_;
    float var = 0.f; for (int c = 0; c < H_; ++c) { float d = SC[c] - mu; var += d * d; }
    var /= H_; RD[2] = mu; RD[3] = rsqrtf(var + 1e-5f);
  }
  __syncthreads();
  float out = (z - RD[2]) * RD[3] * ln_g[tid] + ln_b[tid];
  wres[((size_t)b * N_ + n) * H_ + tid] = (_Float16)out;
}

// ----------------------------------------------------------------- q|k projection
__global__ void qkv_kernel(const _Float16* __restrict__ wres,
    const float* __restrict__ mw, const float* __restrict__ mb,
    _Float16* __restrict__ qk)
{
  extern __shared__ char smem[];
  _Float16* WQ = (_Float16*)smem;   // [256][128] (first 256 rows of mha_in_w)
  _Float16* Xr = WQ + QK * H_;      // [32][128]
  const int tid = threadIdx.x, r0 = blockIdx.x * 32;
  for (int e = tid; e < QK * H_; e += 256) WQ[e] = (_Float16)mw[e];
  for (int e = tid; e < 32 * H_; e += 256) Xr[e] = wres[(size_t)r0 * H_ + e];
  __syncthreads();
  const int w = tid >> 5, lane = tid & 31;
  for (int tt = w; tt < 2 * (QK / 16); tt += 8) {
    int mt = tt & 1, nt = tt >> 1;
    v8f acc = {};
    #pragma unroll
    for (int kk = 0; kk < 4; ++kk)
      acc = wmma16(frag_a(Xr + mt * 16 * H_ + kk * 32, H_),
                   frag_b(WQ + nt * 16 * H_ + kk * 32, H_), acc);
    int col = nt * 16 + (lane & 15), rb = mt * 16 + ((lane >> 4) << 3);
    float bb = mb[col];
    #pragma unroll
    for (int r = 0; r < 8; ++r)
      qk[(size_t)(r0 + rb + r) * QK + col] = (_Float16)(acc[r] + bb);
  }
}

// ------------------------------------------ MHA scores -> softmax -> mean accum
__global__ void scores_kernel(const _Float16* __restrict__ qk,
                              float* __restrict__ attn_acc)
{
  extern __shared__ char smem[];
  _Float16* Qs = (_Float16*)smem;     // [16][32]  (dh=16 padded to K=32)
  _Float16* Ks = Qs + 16 * 32;        // [304][32]
  float* Ss = (float*)(Ks + NP * 32); // [16][304]
  const int strip = blockIdx.x;
  const int bh = blockIdx.y, b = bh >> 3, h = bh & 7;
  const int tid = threadIdx.x;
  for (int e = tid; e < 16 * 32; e += 128) {
    int r = e >> 5, d = e & 31; int n = strip * 16 + r;
    _Float16 v = (_Float16)0.f;
    if (n < N_ && d < 16) v = qk[((size_t)b * N_ + n) * QK + h * 16 + d];
    Qs[e] = v;
  }
  for (int e = tid; e < NP * 32; e += 128) {
    int m = e >> 5, d = e & 31;
    _Float16 v = (_Float16)0.f;
    if (m < N_ && d < 16) v = qk[((size_t)b * N_ + m) * QK + 128 + h * 16 + d];
    Ks[e] = v;
  }
  __syncthreads();
  const int w = tid >> 5, lane = tid & 31;
  for (int nt = w; nt < NP / 16; nt += 4) {
    v8f acc = {};
    acc = wmma16(frag_a(Qs, 32), frag_b(Ks + nt * 16 * 32, 32), acc);
    int col = nt * 16 + (lane & 15), rb = ((lane >> 4) << 3);
    #pragma unroll
    for (int r = 0; r < 8; ++r) Ss[(rb + r) * NP + col] = acc[r] * 0.25f; // /sqrt(16)
  }
  __syncthreads();
  if (tid < 16) {
    int n = strip * 16 + tid;
    if (n < N_) {
      float m = -1e30f;
      for (int j = 0; j < N_; ++j) m = fmaxf(m, Ss[tid * NP + j]);
      float s = 0.f;
      for (int j = 0; j < N_; ++j) { float e = __expf(Ss[tid * NP + j] - m); Ss[tid * NP + j] = e; s += e; }
      float inv = 1.f / (s * 256.f);            // mean over B*NH folded in
      for (int j = 0; j < N_; ++j) Ss[tid * NP + j] *= inv;
    }
  }
  __syncthreads();
  for (int e = tid; e < 16 * N_; e += 128) {
    int r = e / N_, j = e % N_; int n = strip * 16 + r;
    if (n < N_) atomicAdd(&attn_acc[n * N_ + j], Ss[r * NP + j]);
  }
}

// -------------------------------------------------- semantic embed + graph algebra
__global__ void te_kernel(const float* __restrict__ se, const float* __restrict__ sw,
    const float* __restrict__ sb, float* __restrict__ te, float* __restrict__ nrm)
{
  __shared__ float sq[SEM_DIM];
  const int n = blockIdx.x, j = threadIdx.x;
  float acc = sb[j];
  for (int k = 0; k < SEM_IN; ++k) acc += se[n * SEM_IN + k] * sw[k * SEM_DIM + j];
  te[n * SEM_DIM + j] = acc;
  sq[j] = acc * acc;
  __syncthreads();
  if (j == 0) { float s = 0.f; for (int k = 0; k < SEM_DIM; ++k) s += sq[k]; nrm[n] = sqrtf(s); }
}

__global__ void adj_kernel(const float* __restrict__ te, const float* __restrict__ nrm,
    const float* __restrict__ dist, const float* __restrict__ attn,
    const float* __restrict__ alpha_p, float* __restrict__ A, float* __restrict__ A_out)
{
  __shared__ float ti[SEM_DIM];
  __shared__ float MJ[320];
  __shared__ float rowmax;
  const int i = blockIdx.x, j = threadIdx.x;
  for (int k = j; k < SEM_DIM; k += 320) ti[k] = te[i * SEM_DIM + k];
  __syncthreads();
  float mix = 0.f;
  if (j < N_) {
    float d = 0.f;
    for (int k = 0; k < SEM_DIM; ++k) d += ti[k] * te[j * SEM_DIM + k];
    float cs = fabsf(d / (nrm[i] * nrm[j]));
    float al = alpha_p[0];
    mix = (al * dist[i * N_ + j] + (1.f - al) * cs) * attn[i * N_ + j];
    MJ[j] = mix;
  }
  __syncthreads();
  if (j == 0) { float m = -1e30f; for (int k = 0; k < N_; ++k) m = fmaxf(m, MJ[k]); rowmax = m; }
  __syncthreads();
  if (j < N_) {
    float ratio = mix / rowmax;
    float av = (__powf(ratio, 2.5f) > 0.15f) ? mix : 0.f;
    A[i * N_ + j] = av;
    A_out[i * N_ + j] = av;
  }
}

__global__ void deg_kernel(const float* __restrict__ A, float* __restrict__ dinv)
{
  int j = threadIdx.x;
  if (j >= N_) return;
  float s = 0.f;
  for (int i = 0; i < N_; ++i) s += A[i * N_ + j];
  if (A[j * N_ + j] == 0.f) s += 1.f;            // add_remaining_self_loops
  dinv[j] = (s > 0.f) ? rsqrtf(s) : 0.f;
}

__global__ void anorm_kernel(const float* __restrict__ A, const float* __restrict__ dinv,
                             _Float16* __restrict__ AnT)
{
  int i = blockIdx.x;       // 0..303 (row of A_norm^T)
  int j = threadIdx.x;      // 0..319 (K dim, padded)
  float v = 0.f;
  if (i < N_ && j < N_) {
    float ah = A[j * N_ + i] + ((i == j && A[j * N_ + j] == 0.f) ? 1.f : 0.f);
    v = dinv[j] * ah * dinv[i];
  }
  AnT[(size_t)i * JP + j] = (_Float16)v;
}

// ----------------------------------------------------- node features Xf = [Xt|te]
__global__ void fcta_kernel(const _Float16* __restrict__ wres,
    const float* __restrict__ fw, const float* __restrict__ fb,
    const float* __restrict__ te, _Float16* __restrict__ Xf)
{
  extern __shared__ char smem[];
  _Float16* WT = (_Float16*)smem;      // [176][128]  WT[j][k] = fcta_w[k][j]
  _Float16* Xr = WT + TP * H_;         // [32][128]
  const int tid = threadIdx.x, r0 = blockIdx.x * 32;
  for (int e = tid; e < TP * H_; e += 256) {
    int j = e >> 7, k = e & 127;
    WT[e] = (j < T_) ? (_Float16)fw[k * T_ + j] : (_Float16)0.f;
  }
  for (int e = tid; e < 32 * H_; e += 256) Xr[e] = wres[(size_t)r0 * H_ + e];
  __syncthreads();
  const int w = tid >> 5, lane = tid & 31;
  for (int tt = w; tt < 2 * (TP / 16); tt += 8) {
    int mt = tt & 1, nt = tt >> 1;
    v8f acc = {};
    #pragma unroll
    for (int kk = 0; kk < 4; ++kk)
      acc = wmma16(frag_a(Xr + mt * 16 * H_ + kk * 32, H_),
                   frag_b(WT + nt * 16 * H_ + kk * 32, H_), acc);
    int col = nt * 16 + (lane & 15), rb = mt * 16 + ((lane >> 4) << 3);
    if (col < T_) {
      float bb = fb[col];
      #pragma unroll
      for (int r = 0; r < 8; ++r) {
        float v = acc[r] + bb; v = v > 0.f ? v : 0.f;
        Xf[(size_t)(r0 + rb + r) * FNP + col] = (_Float16)v;
      }
    }
  }
  for (int e = tid; e < 32 * SEM_DIM; e += 256) {
    int r = e / SEM_DIM, jj = e % SEM_DIM;
    int gr = r0 + r, nn = gr % N_;
    Xf[(size_t)gr * FNP + T_ + jj] = (_Float16)te[nn * SEM_DIM + jj];
  }
  for (int e = tid; e < 32 * (FNP - FN); e += 256) {
    int r = e / (FNP - FN), jj = e % (FNP - FN);
    Xf[(size_t)(r0 + r) * FNP + FN + jj] = (_Float16)0.f;
  }
}

// ------------------------------------------------------------------ GCN: X @ W
__global__ void xw_kernel(const _Float16* __restrict__ Xin, int kpad, int kreal,
    const float* __restrict__ Wg, _Float16* __restrict__ XW)
{
  extern __shared__ char smem[];
  _Float16* WT = (_Float16*)smem;       // [64][kpad]  WT[c][k] = Wg[k][c]
  _Float16* Xr = WT + CH * kpad;        // [32][kpad]
  const int tid = threadIdx.x, r0 = blockIdx.x * 32;
  for (int e = tid; e < CH * kpad; e += 256) {
    int c = e / kpad, k = e % kpad;
    WT[e] = (k < kreal) ? (_Float16)Wg[k * CH + c] : (_Float16)0.f;
  }
  for (int e = tid; e < 32 * kpad; e += 256) Xr[e] = Xin[(size_t)r0 * kpad + e];
  __syncthreads();
  const int w = tid >> 5, lane = tid & 31;
  for (int tt = w; tt < 2 * (CH / 16); tt += 8) {
    int mt = tt & 1, nt = tt >> 1;
    v8f acc = {};
    for (int kk = 0; kk < kpad / 32; ++kk)
      acc = wmma16(frag_a(Xr + mt * 16 * kpad + kk * 32, kpad),
                   frag_b(WT + nt * 16 * kpad + kk * 32, kpad), acc);
    int col = nt * 16 + (lane & 15), rb = mt * 16 + ((lane >> 4) << 3);
    #pragma unroll
    for (int r = 0; r < 8; ++r) {
      int gr = r0 + rb + r; int bb = gr / N_, nn = gr % N_;
      XW[((size_t)bb * NP + nn) * CH + col] = (_Float16)acc[r];
    }
  }
}

// ------------------------------------------------- GCN: A_norm^T @ XW (+b, relu)
__global__ void agg_kernel(const _Float16* __restrict__ AnT,
    const _Float16* __restrict__ XW, const float* __restrict__ bias,
    _Float16* __restrict__ Xg)
{
  extern __shared__ char smem[];
  _Float16* As = (_Float16*)smem;      // [16][320]
  _Float16* XT = As + 16 * JP;         // [64][320]  XT[f][j] = XW[b][j][f]
  const int strip = blockIdx.x, b = blockIdx.y, tid = threadIdx.x;
  for (int e = tid; e < 16 * JP; e += 128) As[e] = AnT[(size_t)strip * 16 * JP + e];
  for (int e = tid; e < CH * JP; e += 128) {
    int f = e / JP, j = e % JP;
    XT[e] = (j < NP) ? XW[((size_t)b * NP + j) * CH + f] : (_Float16)0.f;
  }
  __syncthreads();
  const int w = tid >> 5, lane = tid & 31;
  for (int nt = w; nt < CH / 16; nt += 4) {
    v8f acc = {};
    #pragma unroll
    for (int kk = 0; kk < JP / 32; ++kk)
      acc = wmma16(frag_a(As + kk * 32, JP), frag_b(XT + nt * 16 * JP + kk * 32, JP), acc);
    int col = nt * 16 + (lane & 15), rb = ((lane >> 4) << 3);
    float bb = bias[col];
    #pragma unroll
    for (int r = 0; r < 8; ++r) {
      int i = strip * 16 + rb + r;
      if (i < N_) {
        float v = acc[r] + bb; v = v > 0.f ? v : 0.f;
        Xg[((size_t)b * N_ + i) * CH + col] = (_Float16)v;
      }
    }
  }
}

// -------------------------------------------------------------------- readout
__global__ void readout_kernel(const _Float16* __restrict__ Xg,
    const float* __restrict__ w1, const float* __restrict__ b1,
    const float* __restrict__ w2, const float* __restrict__ b2,
    float* __restrict__ out)
{
  extern __shared__ char smem[];
  _Float16* W1 = (_Float16*)smem;      // [336][64]  W1[j][k] = fc1_w[k][j]
  _Float16* Xr = W1 + FN * CH;         // [32][64]
  _Float16* Hs = Xr + 32 * CH;         // [32][352]  leaky(fc1) output, K padded
  _Float16* W2 = Hs + 32 * FNP;        // [16][352]  W2[c][k] = fc2_w[k][c]
  const int tid = threadIdx.x, r0 = blockIdx.x * 32;
  for (int e = tid; e < FN * CH; e += 256) { int j = e >> 6, k = e & 63; W1[e] = (_Float16)w1[k * FN + j]; }
  for (int e = tid; e < 32 * CH; e += 256) Xr[e] = Xg[(size_t)r0 * CH + e];
  for (int e = tid; e < 16 * FNP; e += 256) {
    int c = e / FNP, k = e % FNP;
    W2[e] = (c < HOR && k < FN) ? (_Float16)w2[k * HOR + c] : (_Float16)0.f;
  }
  __syncthreads();
  const int w = tid >> 5, lane = tid & 31;
  for (int tt = w; tt < 2 * (FN / 16); tt += 8) {
    int mt = tt & 1, nt = tt >> 1;
    v8f acc = {};
    #pragma unroll
    for (int kk = 0; kk < 2; ++kk)
      acc = wmma16(frag_a(Xr + mt * 16 * CH + kk * 32, CH),
                   frag_b(W1 + nt * 16 * CH + kk * 32, CH), acc);
    int col = nt * 16 + (lane & 15), rb = mt * 16 + ((lane >> 4) << 3);
    float bb = b1[col];
    #pragma unroll
    for (int r = 0; r < 8; ++r) {
      float v = acc[r] + bb; v = v > 0.f ? v : 0.01f * v;   // LeakyReLU
      Hs[(rb + r) * FNP + col] = (_Float16)v;
    }
  }
  for (int e = tid; e < 32 * (FNP - FN); e += 256) {
    int r = e / (FNP - FN), k = e % (FNP - FN);
    Hs[r * FNP + FN + k] = (_Float16)0.f;
  }
  __syncthreads();
  for (int tt = w; tt < 2; tt += 8) {
    int mt = tt;
    v8f acc = {};
    for (int kk = 0; kk < FNP / 32; ++kk)
      acc = wmma16(frag_a(Hs + mt * 16 * FNP + kk * 32, FNP), frag_b(W2 + kk * 32, FNP), acc);
    int col = (lane & 15), rb = mt * 16 + ((lane >> 4) << 3);
    if (col < HOR) {
      float bb = b2[col];
      #pragma unroll
      for (int r = 0; r < 8; ++r) {
        int gr = r0 + rb + r; int b = gr / N_, nn = gr % N_;
        out[((size_t)b * HOR + col) * N_ + nn] = acc[r] + bb;  // [B,HOR,N]
      }
    }
  }
}

// ====================================================================== launch
extern "C" void kernel_launch(void* const* d_in, const int* in_sizes, int n_in,
                              void* d_out, int out_size, void* d_ws, size_t ws_size,
                              hipStream_t stream)
{
  (void)in_sizes; (void)n_in; (void)out_size; (void)ws_size;
  const float* x    = (const float*)d_in[0];
  const float* dist = (const float*)d_in[1];
  const float* sem  = (const float*)d_in[2];
  const float* s1w  = (const float*)d_in[3];
  const float* s1b  = (const float*)d_in[4];
  const float* s2w  = (const float*)d_in[5];
  const float* s2b  = (const float*)d_in[6];
  const float* gwih = (const float*)d_in[7];
  const float* gwhh = (const float*)d_in[8];
  const float* gbih = (const float*)d_in[9];
  const float* gbhh = (const float*)d_in[10];
  const float* aw   = (const float*)d_in[11];
  const float* ab   = (const float*)d_in[12];
  const float* av   = (const float*)d_in[13];
  const float* lng  = (const float*)d_in[14];
  const float* lnb  = (const float*)d_in[15];
  const float* mw   = (const float*)d_in[16];
  const float* mb   = (const float*)d_in[17];
  const float* fw   = (const float*)d_in[18];
  const float* fb   = (const float*)d_in[19];
  const float* sw   = (const float*)d_in[20];
  const float* sb   = (const float*)d_in[21];
  const float* alpha= (const float*)d_in[22];
  const float* g1w = (const float*)d_in[23]; const float* g1b = (const float*)d_in[24];
  const float* g2w = (const float*)d_in[25]; const float* g2b = (const float*)d_in[26];
  const float* g3w = (const float*)d_in[27]; const float* g3b = (const float*)d_in[28];
  const float* f1w = (const float*)d_in[29]; const float* f1b = (const float*)d_in[30];
  const float* f2w = (const float*)d_in[31]; const float* f2b = (const float*)d_in[32];
  float* out  = (float*)d_out;
  float* outA = out + (size_t)B_ * HOR * N_;

  char* p = (char*)d_ws;
  auto take = [&](size_t bytes) {
    char* q = p; p += (bytes + 255) & ~size_t(255); return q;
  };
  _Float16* gru16  = (_Float16*)take((size_t)N_ * B_ * T_ * H_ * 2);
  float*    hid32  = (float*)   take((size_t)N_ * B_ * H_ * 4);
  _Float16* wres16 = (_Float16*)take((size_t)B_ * N_ * H_ * 2);
  _Float16* qk16   = (_Float16*)take((size_t)B_ * N_ * QK * 2);
  float*    attnNN = (float*)   take((size_t)N_ * N_ * 4);
  float*    te32   = (float*)   take((size_t)N_ * SEM_DIM * 4);
  float*    nrm    = (float*)   take(N_ * 4);
  float*    Abuf   = (float*)   take((size_t)N_ * N_ * 4);
  float*    dinv   = (float*)   take(N_ * 4);
  _Float16* AnT16  = (_Float16*)take((size_t)NP * JP * 2);
  _Float16* Xf16   = (_Float16*)take((size_t)B_ * N_ * FNP * 2);
  _Float16* XW16   = (_Float16*)take((size_t)B_ * NP * CH * 2);
  _Float16* XgA    = (_Float16*)take((size_t)B_ * N_ * CH * 2);
  _Float16* XgB    = (_Float16*)take((size_t)B_ * N_ * CH * 2);

  hipMemsetAsync(attnNN, 0, (size_t)N_ * N_ * 4, stream);

  size_t ldsG = (size_t)(G3 * 64 + G3 * H_ + B_ * H_ + B_ * 64) * 2
              + (size_t)(B_ * H_ + 2 * B_ * G3 + 32 + 32 + 32 * 64 + 64 + 2 * G3) * 4;
  gru_kernel<<<N_, 256, ldsG, stream>>>(x, gwih, gwhh, gbih, gbhh,
                                        s1w, s1b, s2w, s2b, gru16, hid32);

  size_t ldsA = (size_t)(H_ * 256 + TP * H_) * 2 + (size_t)(3 * H_ + TP + 4) * 4;
  attn_time_kernel<<<dim3(N_, B_), 128, ldsA, stream>>>(gru16, hid32, aw, ab, av,
                                                        lng, lnb, wres16);

  size_t ldsQ = (size_t)(QK * H_ + 32 * H_) * 2;
  qkv_kernel<<<(B_ * N_) / 32, 256, ldsQ, stream>>>(wres16, mw, mb, qk16);

  size_t ldsS = (size_t)(16 * 32 + NP * 32) * 2 + (size_t)(16 * NP) * 4;
  scores_kernel<<<dim3(NP / 16, B_ * NH_), 128, ldsS, stream>>>(qk16, attnNN);

  te_kernel<<<N_, SEM_DIM, 0, stream>>>(sem, sw, sb, te32, nrm);
  adj_kernel<<<N_, 320, 0, stream>>>(te32, nrm, dist, attnNN, alpha, Abuf, outA);
  deg_kernel<<<1, 320, 0, stream>>>(Abuf, dinv);
  anorm_kernel<<<NP, JP, 0, stream>>>(Abuf, dinv, AnT16);

  size_t lds7 = (size_t)(TP * H_ + 32 * H_) * 2;
  fcta_kernel<<<(B_ * N_) / 32, 256, lds7, stream>>>(wres16, fw, fb, te32, Xf16);

  size_t lds8a1 = (size_t)(CH * FNP + 32 * FNP) * 2;
  size_t lds8a2 = (size_t)(CH * CH + 32 * CH) * 2;
  size_t lds8b  = (size_t)(16 * JP + CH * JP) * 2;
  size_t xwBytes = (size_t)B_ * NP * CH * 2;

  hipMemsetAsync(XW16, 0, xwBytes, stream);
  xw_kernel<<<(B_ * N_) / 32, 256, lds8a1, stream>>>(Xf16, FNP, FN, g1w, XW16);
  agg_kernel<<<dim3(NP / 16, B_), 128, lds8b, stream>>>(AnT16, XW16, g1b, XgA);

  hipMemsetAsync(XW16, 0, xwBytes, stream);
  xw_kernel<<<(B_ * N_) / 32, 256, lds8a2, stream>>>(XgA, CH, CH, g2w, XW16);
  agg_kernel<<<dim3(NP / 16, B_), 128, lds8b, stream>>>(AnT16, XW16, g2b, XgB);

  hipMemsetAsync(XW16, 0, xwBytes, stream);
  xw_kernel<<<(B_ * N_) / 32, 256, lds8a2, stream>>>(XgB, CH, CH, g3w, XW16);
  agg_kernel<<<dim3(NP / 16, B_), 128, lds8b, stream>>>(AnT16, XW16, g3b, XgA);

  size_t lds9 = (size_t)(FN * CH + 32 * CH + 32 * FNP + 16 * FNP) * 2;
  readout_kernel<<<(B_ * N_) / 32, 256, lds9, stream>>>(XgA, f1w, f1b, f2w, f2b, out);
}
